// TinyDiffusionModelHardware_39926015984304
// MI455X (gfx1250) — matrix operand order repaired and checked by reading, not compiled, not run on hardware
//
#include <hip/hip_runtime.h>
#include <hip/hip_bf16.h>

typedef __attribute__((ext_vector_type(16))) _Float16 v16h;
typedef __attribute__((ext_vector_type(8)))  _Float16 v8h;
typedef __attribute__((ext_vector_type(8)))  float    v8f;
typedef __attribute__((ext_vector_type(4)))  float    v4f;
typedef __attribute__((ext_vector_type(4)))  int      v4i;

#define BATCH      4096
#define LATENT     20480     // FRAME_COUNT*32*32
#define HIDDEN     4096
#define TAILDIM    128       // COND_DIM + TIME_DIM
#define IN_DIM     20608     // LATENT + TAILDIM

#define AS1 __attribute__((address_space(1)))
#define AS3 __attribute__((address_space(3)))

#if defined(__HIP_DEVICE_COMPILE__) && \
    __has_builtin(__builtin_amdgcn_global_load_async_to_lds_b128)
#define ASYNC_A 1
#else
#define ASYNC_A 0
#endif

__device__ __forceinline__ void wait_async_lds() {
#if ASYNC_A
#if __has_builtin(__builtin_amdgcn_s_wait_asynccnt)
    __builtin_amdgcn_s_wait_asynccnt(0);
#else
    asm volatile("s_wait_asynccnt 0x0" ::: "memory");
#endif
#endif
}

// ---------------------------------------------------------------------------
// tail[b,0:64] = emb_table[digits[b]];  tail[b,64:128] = time MLP. (fp32)
// ---------------------------------------------------------------------------
__global__ __launch_bounds__(64)
void tail_kernel(const int* __restrict__ digits, const int* __restrict__ timesteps,
                 const int* __restrict__ num_steps_p,
                 const float* __restrict__ emb,
                 const float* __restrict__ tw0, const float* __restrict__ tb0,
                 const float* __restrict__ tw1, const float* __restrict__ tb1,
                 float* __restrict__ tail)
{
    const int b = blockIdx.x;
    const int j = threadIdx.x;           // 0..63
    __shared__ float h[64];

    const int ns = num_steps_p[0];
    const int denom = (ns - 1) > 1 ? (ns - 1) : 1;
    const float t  = (float)timesteps[b] / (float)denom;
    const float pi = 3.14159265358979323846f;
    const float s2 = sinf(pi * t), s3 = cosf(pi * t);

    float a = tb0[j] + t * tw0[j * 4 + 0] + (t * t) * tw0[j * 4 + 1]
                     + s2 * tw0[j * 4 + 2] + s3 * tw0[j * 4 + 3];
    h[j] = fmaxf(a, 0.0f);
    __syncthreads();

    float te = tb1[j];
    #pragma unroll 8
    for (int i = 0; i < 64; ++i) te += h[i] * tw1[j * 64 + i];

    tail[(size_t)b * TAILDIM + j]      = emb[digits[b] * 64 + j];
    tail[(size_t)b * TAILDIM + 64 + j] = te;
}

// ---------------------------------------------------------------------------
// Helpers
// ---------------------------------------------------------------------------
__device__ __forceinline__ v16h cat16(v8h lo, v8h hi) {
    union U { v16h v; v8h h[2]; } u;
    u.h[0] = lo; u.h[1] = hi;
    return u.v;
}

__device__ __forceinline__ void cvt_store16(const v4f* r, _Float16* dst) {
    v8h lo, hi;
    #pragma unroll
    for (int i = 0; i < 4; ++i) {
        lo[i]     = (_Float16)r[0][i];
        lo[i + 4] = (_Float16)r[1][i];
        hi[i]     = (_Float16)r[2][i];
        hi[i + 4] = (_Float16)r[3][i];
    }
    *(v8h*)dst       = lo;
    *(v8h*)(dst + 8) = hi;
}

// ---------------------------------------------------------------------------
// Fuse x (fp32) + tail (fp32) -> contiguous net_in (f16) [BATCH, IN_DIM].
// ---------------------------------------------------------------------------
__global__ __launch_bounds__(256)
void build_netin(const float* __restrict__ x, const float* __restrict__ tail,
                 _Float16* __restrict__ netin)
{
    const size_t idx = ((size_t)blockIdx.x * 256 + threadIdx.x) * 16;
    const size_t row = idx / IN_DIM;
    const int    col = (int)(idx % IN_DIM);
    const float* src = (col < LATENT)
        ? x    + row * LATENT  + col
        : tail + row * TAILDIM + (col - LATENT);
    v4f r[4];
    #pragma unroll
    for (int i = 0; i < 4; ++i) r[i] = *(const v4f*)(src + 4 * i);
    cvt_store16(r, netin + idx);
}

// ---------------------------------------------------------------------------
// GEMM: out[M,N] = act( A[M,K] * W[N,K]^T + bias[N] )
//   Double-buffered LDS, one barrier / K-step, 128x128x32 tile, 256 threads,
//   8 waves (2x4), each wave: 4x2 v_wmma_f32_16x16x32_f16 accumulators.
//   A f16 path staged via GLOBAL_LOAD_ASYNC_TO_LDS_B128 when available.
// ---------------------------------------------------------------------------
template<bool A_HALF, bool SPLIT, bool RELU, bool OUT_HALF>
__global__ __launch_bounds__(256)
void gemm_f16wmma(const float*    __restrict__ Af,
                  const _Float16* __restrict__ Ah,
                  const float*    __restrict__ Atail,
                  const float*    __restrict__ W,
                  const float*    __restrict__ bias,
                  float*          __restrict__ outF,
                  _Float16*       __restrict__ outH,
                  int N, int K, int ldA, int kSplit, int groupM)
{
    constexpr int BM = 128, BN = 128, BK = 32, LSTR = 40;   // 80B LDS row stride
    __shared__ __align__(16) _Float16 As[2][BM][LSTR];
    __shared__ __align__(16) _Float16 Bs[2][BN][LSTR];

    const int t = threadIdx.x;

    // grouped ordering: groupM consecutive M-tiles share each W panel via L2
    const int nTilesN = N / BN;
    const int tilesPerGroup = groupM * nTilesN;
    const int grp = blockIdx.x / tilesPerGroup;
    const int rem = blockIdx.x % tilesPerGroup;
    const int blockM = (grp * groupM + (rem % groupM)) * BM;
    const int blockN = (rem / groupM) * BN;

    const int wave = t >> 5;
    const int lane = t & 31;
    const int wm = (wave & 1) * 64;
    const int wn = (wave >> 1) * 32;
    const int lr = lane & 15;
    const int ls = lane >> 4;

    const int stR = t >> 1;                   // staging row 0..127
    const int stC = (t & 1) * 16;             // staging col 0 or 16

    v8f acc[4][2] = {};

    v4f saf[4];     // A fp32 staging
    v8h sah[2];     // A f16 staging (sync fallback)
    v4f swf[4];     // W fp32 staging

    auto fetch = [&](int k0, int buf) {
        if constexpr (A_HALF) {
            const _Float16* src = Ah + (size_t)(blockM + stR) * ldA + k0 + stC;
#if ASYNC_A
            _Float16* dA = &As[buf][stR][stC];
            __builtin_amdgcn_global_load_async_to_lds_b128(
                (AS1 v4i*)src, (AS3 v4i*)dA, 0, 0);
            __builtin_amdgcn_global_load_async_to_lds_b128(
                (AS1 v4i*)(src + 8), (AS3 v4i*)(dA + 8), 0, 0);
#else
            sah[0] = *(const v8h*)src;
            sah[1] = *(const v8h*)(src + 8);
            (void)buf;
#endif
        } else {
            const float* src;
            if (SPLIT && k0 >= kSplit)
                src = Atail + (size_t)(blockM + stR) * TAILDIM + (k0 - kSplit) + stC;
            else
                src = Af + (size_t)(blockM + stR) * ldA + k0 + stC;
            #pragma unroll
            for (int i = 0; i < 4; ++i) saf[i] = *(const v4f*)(src + 4 * i);
            (void)buf;
        }
        const float* wsrc = W + (size_t)(blockN + stR) * K + k0 + stC;
        #pragma unroll
        for (int i = 0; i < 4; ++i) swf[i] = *(const v4f*)(wsrc + 4 * i);
    };

    auto commit = [&](int buf) {
        if constexpr (A_HALF) {
#if !ASYNC_A
            _Float16* dA = &As[buf][stR][stC];
            *(v8h*)dA       = sah[0];
            *(v8h*)(dA + 8) = sah[1];
#endif
        } else {
            cvt_store16(saf, &As[buf][stR][stC]);
        }
        cvt_store16(swf, &Bs[buf][stR][stC]);
    };

    fetch(0, 0);
    commit(0);
    wait_async_lds();
    __syncthreads();

    const int nK = K / BK;
    for (int kt = 0; kt < nK; ++kt) {
        const int cur = kt & 1;
        const bool more = (kt + 1) < nK;
        if (more) fetch((kt + 1) * BK, cur ^ 1);

        // fragments per ISA 16-bit A/B layouts
        v16h afr[4], bfr[2];
        #pragma unroll
        for (int mi = 0; mi < 4; ++mi) {
            const _Float16* p = &As[cur][wm + mi * 16 + lr][ls * 8];
            afr[mi] = cat16(*(const v8h*)p, *(const v8h*)(p + 16));
        }
        #pragma unroll
        for (int ni = 0; ni < 2; ++ni) {
            const _Float16* p = &Bs[cur][wn + ni * 16 + lr][ls * 8];
            bfr[ni] = cat16(*(const v8h*)p, *(const v8h*)(p + 16));
        }
        #pragma unroll
        for (int mi = 0; mi < 4; ++mi)
            #pragma unroll
            for (int ni = 0; ni < 2; ++ni)
                acc[mi][ni] = __builtin_amdgcn_wmma_f32_16x16x32_f16(
                    false, afr[mi], false, bfr[ni],
                    (short)0, acc[mi][ni], false, false);

        if (more) commit(cur ^ 1);
        wait_async_lds();
        __syncthreads();
    }

    // ---- epilogue: bias (+ReLU); D layout: M = j + 8*ls, N = lr ----
    #pragma unroll
    for (int mi = 0; mi < 4; ++mi) {
        #pragma unroll
        for (int ni = 0; ni < 2; ++ni) {
            const int col = blockN + wn + ni * 16 + lr;
            const float bb = bias[col];
            #pragma unroll
            for (int j = 0; j < 8; ++j) {
                const int rowg = blockM + wm + mi * 16 + ls * 8 + j;
                float v = acc[mi][ni][j] + bb;
                if (RELU) v = fmaxf(v, 0.0f);
                if (OUT_HALF) outH[(size_t)rowg * N + col] = (_Float16)v;
                else          outF[(size_t)rowg * N + col] = v;
            }
        }
    }
}

// ---------------------------------------------------------------------------
// Launch
// ---------------------------------------------------------------------------
extern "C" void kernel_launch(void* const* d_in, const int* in_sizes, int n_in,
                              void* d_out, int out_size, void* d_ws, size_t ws_size,
                              hipStream_t stream)
{
    const float* x   = (const float*)d_in[0];
    const int*   dig = (const int*)  d_in[1];
    const int*   ts  = (const int*)  d_in[2];
    const int*   nsp = (const int*)  d_in[3];
    const float* emb = (const float*)d_in[4];
    const float* tw0 = (const float*)d_in[5];
    const float* tb0 = (const float*)d_in[6];
    const float* tw1 = (const float*)d_in[7];
    const float* tb1 = (const float*)d_in[8];
    const float* nw0 = (const float*)d_in[9];
    const float* nb0 = (const float*)d_in[10];
    const float* nw1 = (const float*)d_in[11];
    const float* nb1 = (const float*)d_in[12];
    const float* nw2 = (const float*)d_in[13];
    const float* nb2 = (const float*)d_in[14];
    float* out = (float*)d_out;

    char* ws = (char*)d_ws;

    const size_t szNetin = (size_t)BATCH * IN_DIM * sizeof(_Float16);   // 168.9 MB
    const size_t szAct   = (size_t)BATCH * HIDDEN * sizeof(_Float16);   // 32 MB
    const size_t szTail  = (size_t)BATCH * TAILDIM * sizeof(float);     // 2 MB
    const bool fast = ws_size >= (szNetin + 2 * szAct + szTail);

    const int gMM = (BATCH / 128) * (HIDDEN / 128);   // 1024
    const int gMO = (BATCH / 128) * (LATENT / 128);   // 5120

    if (fast) {
        _Float16* netin = (_Float16*)ws;
        _Float16* a0    = (_Float16*)(ws + szNetin);
        _Float16* a1    = (_Float16*)(ws + szNetin + szAct);
        float*    tail  = (float*)   (ws + szNetin + 2 * szAct);

        tail_kernel<<<BATCH, 64, 0, stream>>>(dig, ts, nsp, emb, tw0, tb0, tw1, tb1, tail);
        build_netin<<<(int)((size_t)BATCH * IN_DIM / 16 / 256), 256, 0, stream>>>(x, tail, netin);

        gemm_f16wmma<true, false, true, true><<<gMM, 256, 0, stream>>>(
            nullptr, netin, nullptr, nw0, nb0, nullptr, a0,
            HIDDEN, IN_DIM, IN_DIM, 0, BATCH / 128);
        gemm_f16wmma<true, false, true, true><<<gMM, 256, 0, stream>>>(
            nullptr, a0, nullptr, nw1, nb1, nullptr, a1,
            HIDDEN, HIDDEN, HIDDEN, 0, BATCH / 128);
        gemm_f16wmma<true, false, false, false><<<gMO, 256, 0, stream>>>(
            nullptr, a1, nullptr, nw2, nb2, out, nullptr,
            LATENT, HIDDEN, HIDDEN, 0, BATCH / 128);
    } else {
        float*    tail = (float*)ws;
        _Float16* a0   = (_Float16*)(ws + szTail);
        _Float16* a1   = a0 + (size_t)BATCH * HIDDEN;

        tail_kernel<<<BATCH, 64, 0, stream>>>(dig, ts, nsp, emb, tw0, tb0, tw1, tb1, tail);

        gemm_f16wmma<false, true, true, true><<<gMM, 256, 0, stream>>>(
            x, nullptr, tail, nw0, nb0, nullptr, a0,
            HIDDEN, IN_DIM, LATENT, LATENT, 8);
        gemm_f16wmma<true, false, true, true><<<gMM, 256, 0, stream>>>(
            nullptr, a0, nullptr, nw1, nb1, nullptr, a1,
            HIDDEN, HIDDEN, HIDDEN, 0, BATCH / 128);
        gemm_f16wmma<true, false, false, false><<<gMO, 256, 0, stream>>>(
            nullptr, a1, nullptr, nw2, nb2, out, nullptr,
            LATENT, HIDDEN, HIDDEN, 0, BATCH / 128);
    }
}
